// feature_extractor_22127671509737
// MI455X (gfx1250) — matrix operand order repaired and checked by reading, not compiled, not run on hardware
//
#include <hip/hip_runtime.h>
#include <hip/hip_bf16.h>

// ---------------------------------------------------------------------------
// Feature extractor for MI455X (gfx1250, wave32).
// HC block = scalar stencil kernels; 5x5 convs = implicit GEMM on
// v_wmma_f32_16x16x32_f16 with hi/lo f16 splitting for ~fp32 accuracy.
// ---------------------------------------------------------------------------

typedef __attribute__((ext_vector_type(16))) _Float16 v16h;
typedef __attribute__((ext_vector_type(8)))  float    v8f;

constexpr int kNB = 8;
constexpr int kH  = 512;
constexpr int kW  = 512;
constexpr int kHW = kH * kW;            // 262144 = 1<<18
constexpr int kPlaneTot = kNB * kHW;    // 2097152 (one [B,1,H,W] tensor)

// workspace offsets (in floats)
constexpr long OFF_P1    = 16777216;    // conv pong buffer [8,8,512,512]
constexpr long OFF_XHC   = 33554432;    // [8,6,512,512]
constexpr long OFF_STATS = 46137344;    // stats + partials

// stats layout (float index from stats base):
//  uint keys: [0..7] max(har_raw)  [8..15] min(har_b) [16..23] max(har_b)
//             [24..31] min(shi_b)  [32..39] max(shi_b)
//             [40..47] min(heat)   [48..55] max(heat)
//  [80..87] bn_scale  [88..95] bn_shift
//  [128..4223] bn partial sums, [4224..8319] bn partial sumsq
constexpr int ST_SCALE = 80, ST_SHIFT = 88, ST_PSUM = 128, ST_PSSQ = 128 + 4096;

// ---------------- helpers ----------------
__device__ __forceinline__ int clampi(int v, int lo, int hi) {
  return v < lo ? lo : (v > hi ? hi : v);
}
__device__ __forceinline__ int refl(int i, int n) {   // numpy 'reflect'
  if (i < 0) i = -i;
  if (i >= n) i = 2 * n - 2 - i;
  return i;
}
// monotone float<->uint mapping for atomic min/max
__device__ __forceinline__ unsigned fkey(float f) {
  unsigned u = __float_as_uint(f);
  return (u & 0x80000000u) ? ~u : (u | 0x80000000u);
}
__device__ __forceinline__ float funkey(unsigned k) {
  return (k & 0x80000000u) ? __uint_as_float(k & 0x7fffffffu)
                           : __uint_as_float(~k);
}

__constant__ float K7[7] = {0.00443305f, 0.05400558f, 0.24203623f,
                            0.39905028f, 0.24203623f, 0.05400558f, 0.00443305f};
__constant__ float K3[3] = {0.27406862f, 0.45186276f, 0.27406862f};
__constant__ int FAST_DX[16] = {0,1,2,3,3,3,2,1,0,-1,-2,-3,-3,-3,-2,-1};
__constant__ int FAST_DY[16] = {3,3,2,1,0,-1,-2,-3,-3,-3,-2,-1,0,1,2,3};

// ---------------- HC block kernels ----------------
__global__ void k_zero_stats(unsigned* u) {
  int i = threadIdx.x;
  if (i < 56) {
    int blk = i >> 3;                       // 0..6
    bool isMin = (blk == 1) || (blk == 3) || (blk == 5);
    u[i] = isMin ? 0xFFFFFFFFu : 0u;
  }
}

__global__ __launch_bounds__(256) void k_gray(const float* __restrict__ x,
                                              float* __restrict__ g) {
  int idx = blockIdx.x * 256 + threadIdx.x;
  if (idx >= kPlaneTot) return;
  int b = idx >> 18, p = idx & (kHW - 1);
  const float* xb = x + (long)b * 3 * kHW;
  g[idx] = 0.299f * xb[p] + 0.587f * xb[kHW + p] + 0.114f * xb[2 * kHW + p];
}

__global__ __launch_bounds__(256) void k_sobel(const float* __restrict__ g,
                                               float* __restrict__ dx,
                                               float* __restrict__ dy) {
  int idx = blockIdx.x * 256 + threadIdx.x;
  if (idx >= kPlaneTot) return;
  int b = idx >> 18, p = idx & (kHW - 1);
  int y = p >> 9, x = p & 511;
  const float* gb = g + (long)b * kHW;
  auto G = [&](int yy, int xx) {
    return gb[(clampi(yy, 0, 511) << 9) + clampi(xx, 0, 511)];
  };
  float a00 = G(y - 1, x - 1), a01 = G(y - 1, x), a02 = G(y - 1, x + 1);
  float a10 = G(y, x - 1),                         a12 = G(y, x + 1);
  float a20 = G(y + 1, x - 1), a21 = G(y + 1, x), a22 = G(y + 1, x + 1);
  dx[idx] = (a02 - a00 + 2.f * (a12 - a10) + a22 - a20) * 0.125f;
  dy[idx] = (a20 - a00 + 2.f * (a21 - a01) + a22 - a02) * 0.125f;
}

__global__ __launch_bounds__(256) void k_hblur7(const float* __restrict__ dx,
                                                const float* __restrict__ dy,
                                                float* __restrict__ t0,
                                                float* __restrict__ t1,
                                                float* __restrict__ t2) {
  int idx = blockIdx.x * 256 + threadIdx.x;
  if (idx >= kPlaneTot) return;
  int p = idx & (kHW - 1);
  int x = p & 511;
  long rowBase = (long)idx - x;
  float s0 = 0.f, s1 = 0.f, s2 = 0.f;
#pragma unroll
  for (int t = 0; t < 7; ++t) {
    int xi = refl(x + t - 3, 512);
    float a = dx[rowBase + xi], c = dy[rowBase + xi], w = K7[t];
    s0 += w * a * a; s1 += w * c * c; s2 += w * a * c;
  }
  t0[idx] = s0; t1[idx] = s1; t2[idx] = s2;
}

__global__ __launch_bounds__(256) void k_vblur7(const float* __restrict__ t0,
                                                const float* __restrict__ t1,
                                                const float* __restrict__ t2,
                                                float* __restrict__ o0,
                                                float* __restrict__ o1,
                                                float* __restrict__ o2) {
  int idx = blockIdx.x * 256 + threadIdx.x;
  if (idx >= kPlaneTot) return;
  int b = idx >> 18, p = idx & (kHW - 1);
  int y = p >> 9, x = p & 511;
  long base = (long)b * kHW + x;
  float s0 = 0.f, s1 = 0.f, s2 = 0.f;
#pragma unroll
  for (int t = 0; t < 7; ++t) {
    long off = base + ((long)refl(y + t - 3, 512) << 9);
    float w = K7[t];
    s0 += w * t0[off]; s1 += w * t1[off]; s2 += w * t2[off];
  }
  o0[idx] = s0; o1[idx] = s1; o2[idx] = s2;
}

__global__ __launch_bounds__(256) void k_dettr(const float* __restrict__ dx2,
                                               const float* __restrict__ dy2,
                                               const float* __restrict__ dxy,
                                               float* __restrict__ shi_raw,
                                               float* __restrict__ har_raw,
                                               unsigned* __restrict__ su) {
  int b = blockIdx.x >> 10;
  int off = ((blockIdx.x & 1023) << 8) + threadIdx.x;
  int idx = (b << 18) + off;
  float a = dx2[idx], c = dy2[idx], m = dxy[idx];
  float det = a * c - m * m;
  float tr = a + c;
  shi_raw[idx] = 0.5f * (tr - sqrtf(fabsf(tr * tr - 4.f * det) + 1e-12f));
  float har = det - 0.04f * tr * tr;
  har_raw[idx] = har;
  __shared__ unsigned sm[256];
  sm[threadIdx.x] = fkey(har);
  __syncthreads();
  for (int s = 128; s > 0; s >>= 1) {
    if (threadIdx.x < s) sm[threadIdx.x] = max(sm[threadIdx.x], sm[threadIdx.x + s]);
    __syncthreads();
  }
  if (threadIdx.x == 0) atomicMax(&su[b], sm[0]);
}

// generic 3x3 gaussian (reflect pad) + per-image min/max reduction.
// doThresh: input thresholded at 0.01*max(har_raw) (read from su[0..7]).
__global__ __launch_bounds__(256) void k_blur3(const float* __restrict__ in,
                                               float* __restrict__ out,
                                               unsigned* __restrict__ su,
                                               int minSlot, int maxSlot,
                                               int doThresh) {
  int b = blockIdx.x >> 10;
  int off = ((blockIdx.x & 1023) << 8) + threadIdx.x;
  int y = off >> 9, x = off & 511;
  const float* ib = in + (long)b * kHW;
  float thr = doThresh ? 0.01f * funkey(su[b]) : 0.f;
  float s = 0.f;
#pragma unroll
  for (int i = 0; i < 3; ++i) {
    int yy = refl(y + i - 1, 512);
#pragma unroll
    for (int j = 0; j < 3; ++j) {
      int xx = refl(x + j - 1, 512);
      float v = ib[(yy << 9) + xx];
      if (doThresh) v = (v > thr) ? v : 0.f;
      s += K3[i] * K3[j] * v;
    }
  }
  out[(long)b * kHW + off] = s;
  __shared__ unsigned smn[256], smx[256];
  unsigned k = fkey(s);
  smn[threadIdx.x] = k; smx[threadIdx.x] = k;
  __syncthreads();
  for (int st = 128; st > 0; st >>= 1) {
    if (threadIdx.x < st) {
      smn[threadIdx.x] = min(smn[threadIdx.x], smn[threadIdx.x + st]);
      smx[threadIdx.x] = max(smx[threadIdx.x], smx[threadIdx.x + st]);
    }
    __syncthreads();
  }
  if (threadIdx.x == 0) {
    atomicMin(&su[minSlot + b], smn[0]);
    atomicMax(&su[maxSlot + b], smx[0]);
  }
}

__global__ __launch_bounds__(256) void k_fast(const float* __restrict__ g,
                                              float* __restrict__ score) {
  int idx = blockIdx.x * 256 + threadIdx.x;
  if (idx >= kPlaneTot) return;
  int b = idx >> 18, p = idx & (kHW - 1);
  int y = p >> 9, x = p & 511;
  const float* gb = g + (long)b * kHW;
  float g0 = floorf(gb[p] * 255.f);
  unsigned mb = 0, md = 0;
  float sb = 0.f, sd = 0.f;
#pragma unroll
  for (int i = 0; i < 16; ++i) {
    int yy = clampi(y + FAST_DY[i], 0, 511);
    int xx = clampi(x + FAST_DX[i], 0, 511);
    float nv = floorf(gb[(yy << 9) + xx] * 255.f);
    if (nv > g0 + 10.f) mb |= (1u << i);
    if (nv < g0 - 10.f) md |= (1u << i);
    sb += fmaxf(nv - g0 - 10.f, 0.f);
    sd += fmaxf(g0 - nv - 10.f, 0.f);
  }
  unsigned eb = mb | (mb << 16), ed = md | (md << 16);
  bool arc = false;
#pragma unroll
  for (int s = 0; s < 16; ++s) {
    if (((eb >> s) & 0x1FFu) == 0x1FFu) arc = true;
    if (((ed >> s) & 0x1FFu) == 0x1FFu) arc = true;
  }
  bool border = (y >= 3) && (y < 509) && (x >= 3) && (x < 509);
  score[idx] = (arc && border) ? fmaxf(sb, sd) : 0.f;
}

__global__ __launch_bounds__(256) void k_nms(const float* __restrict__ score,
                                             float* __restrict__ mask) {
  int idx = blockIdx.x * 256 + threadIdx.x;
  if (idx >= kPlaneTot) return;
  int b = idx >> 18, p = idx & (kHW - 1);
  int y = p >> 9, x = p & 511;
  const float* sc = score + (long)b * kHW;
  float v = sc[p];
  float mx = -3.4e38f;
  for (int i = -1; i <= 1; ++i) {
    int yy = y + i;
    if (yy < 0 || yy > 511) continue;
    for (int j = -1; j <= 1; ++j) {
      int xx = x + j;
      if (xx < 0 || xx > 511) continue;
      mx = fmaxf(mx, sc[(yy << 9) + xx]);
    }
  }
  mask[idx] = (v > 0.f && v >= mx) ? 1.f : 0.f;
}

__global__ __launch_bounds__(256) void k_xhc(const float* __restrict__ x,
                                             const float* __restrict__ har_b,
                                             const float* __restrict__ shi_b,
                                             const float* __restrict__ heat,
                                             const unsigned* __restrict__ su,
                                             float* __restrict__ xhc) {
  int idx = blockIdx.x * 256 + threadIdx.x;
  if (idx >= kPlaneTot) return;
  int b = idx >> 18, p = idx & (kHW - 1);
  const float* xb = x + (long)b * 3 * kHW;
  float* ob = xhc + (long)b * 6 * kHW;
  ob[p] = xb[p];
  ob[kHW + p] = xb[kHW + p];
  ob[2 * kHW + p] = xb[2 * kHW + p];
  {
    float mn = funkey(su[8 + b]), mx = funkey(su[16 + b]);
    ob[3 * kHW + p] = (har_b[idx] - mn) / (mx - mn + 1e-8f);
  }
  {
    float mn = funkey(su[24 + b]), mx = funkey(su[32 + b]);
    ob[4 * kHW + p] = (shi_b[idx] - mn) / (mx - mn + 1e-8f);
  }
  {
    float mn = funkey(su[40 + b]), mx = funkey(su[48 + b]);
    float rng = mx - mn, h = heat[idx];
    ob[5 * kHW + p] = (rng > 1e-8f) ? (h - mn) / rng : h;
  }
}

// ---------------- conv 5x5 (pad 2) via WMMA implicit GEMM ----------------
// One wave computes D[16 pixels x 16 oc] (8 real oc). K = Cin*25 padded to
// 32*nChunks. f32 inputs split into f16 hi+lo; acc += Ah*Bh + Ah*Bl + Al*Bh.
__global__ __launch_bounds__(256) void k_conv_wmma(
    const float* __restrict__ in, const float* __restrict__ wgt,
    const float* __restrict__ bias, float* __restrict__ out,
    int Cin, int nChunks) {
  const int lane = threadIdx.x & 31;
  const int wave = threadIdx.x >> 5;
  const int tile = blockIdx.x * 8 + wave;     // 131072 tiles total
  const int b = tile >> 14;                   // / (512*32)
  const int rem = tile & 16383;
  const int y = rem >> 5;
  const int x0 = (rem & 31) << 4;
  const int Kmax = Cin * 25;
  const int m = lane & 15;                    // A row (pixel within tile)
  const int n = lane & 15;                    // B col (output channel)
  const bool hi = lane >= 16;

  v8f acc = {};
  for (int kc = 0; kc < nChunks; ++kc) {
    v16h ah, al, bh, bl;
#pragma unroll
    for (int i = 0; i < 16; ++i) {
      // ---- A fragment: 16-bit A 16x32 layout ----
      int ka = kc * 32 + (hi ? (i < 8 ? i + 8 : i + 16)
                             : (i < 8 ? i : i + 8));
      float av = 0.f;
      if (ka < Kmax) {
        int ic = ka / 25, r = ka % 25, ky = r / 5, kx = r % 5;
        int yy = y + ky - 2, xx = x0 + m + kx - 2;
        if (yy >= 0 && yy < 512 && xx >= 0 && xx < 512)
          av = in[(((long)b * Cin + ic) << 18) + (yy << 9) + xx];
      }
      _Float16 h0 = (_Float16)av;
      ah[i] = h0;
      al[i] = (_Float16)(av - (float)h0);
      // ---- B fragment: 16-bit B 32x16 layout ----
      int kb = kc * 32 + (hi ? (i < 8 ? i + 8 : i + 16) : (i < 8 ? i : i + 8));
      float bv = 0.f;
      if (kb < Kmax && n < 8) {
        int ic = kb / 25, r = kb % 25, ky = r / 5, kx = r % 5;
        bv = wgt[((n * Cin + ic) * 5 + ky) * 5 + kx];
      }
      _Float16 h1 = (_Float16)bv;
      bh[i] = h1;
      bl[i] = (_Float16)(bv - (float)h1);
    }
    acc = __builtin_amdgcn_wmma_f32_16x16x32_f16(false, ah, false, bh,
                                                 (short)0, acc, false, false);
    acc = __builtin_amdgcn_wmma_f32_16x16x32_f16(false, ah, false, bl,
                                                 (short)0, acc, false, false);
    acc = __builtin_amdgcn_wmma_f32_16x16x32_f16(false, al, false, bh,
                                                 (short)0, acc, false, false);
  }
  if (n < 8) {
    float bs = bias[n];
    long base = (((long)b * 8 + n) << 18) + (y << 9) + x0 + (hi ? 8 : 0);
#pragma unroll
    for (int r = 0; r < 8; ++r) out[base + r] = acc[r] + bs;
  }
}

// ---------------- batchnorm (train-mode, biased var) ----------------
// stage 1: per-block partial sum/sumsq, fixed slots (deterministic)
__global__ __launch_bounds__(256) void k_bnstats(const float* __restrict__ y,
                                                 float* __restrict__ psum,
                                                 float* __restrict__ pssq) {
  int plane = blockIdx.x >> 6;                // b*8 + c
  long base = ((long)plane << 18) + ((long)(blockIdx.x & 63) << 12);
  float s = 0.f, ss = 0.f;
#pragma unroll
  for (int e = 0; e < 16; ++e) {
    float v = y[base + e * 256 + threadIdx.x];
    s += v; ss += v * v;
  }
  __shared__ float a[256], c2[256];
  a[threadIdx.x] = s; c2[threadIdx.x] = ss;
  __syncthreads();
  for (int st = 128; st > 0; st >>= 1) {
    if (threadIdx.x < st) {
      a[threadIdx.x] += a[threadIdx.x + st];
      c2[threadIdx.x] += c2[threadIdx.x + st];
    }
    __syncthreads();
  }
  if (threadIdx.x == 0) { psum[blockIdx.x] = a[0]; pssq[blockIdx.x] = c2[0]; }
}

// stage 2: fixed-order final reduce -> scale/shift
__global__ void k_bnfinal(const float* __restrict__ psum,
                          const float* __restrict__ pssq,
                          const float* __restrict__ gamma,
                          const float* __restrict__ beta,
                          float* __restrict__ scale,
                          float* __restrict__ shift) {
  int c = threadIdx.x;
  if (c >= 8) return;
  float s = 0.f, ss = 0.f;
  for (int b = 0; b < 8; ++b)
    for (int j = 0; j < 64; ++j) {
      int p = ((b * 8 + c) << 6) + j;
      s += psum[p]; ss += pssq[p];
    }
  const float N = 2097152.f;
  float mu = s / N;
  float var = ss / N - mu * mu;
  float sc = gamma[c] * rsqrtf(var + 1e-5f);
  scale[c] = sc;
  shift[c] = beta[c] - mu * sc;
}

__global__ __launch_bounds__(256) void k_bnrelu(float* __restrict__ y,
                                                const float* __restrict__ scale,
                                                const float* __restrict__ shift) {
  int idx = blockIdx.x * 256 + threadIdx.x;   // 16777216 elements
  int c = (idx >> 18) & 7;
  float v = y[idx] * scale[c] + shift[c];
  y[idx] = fmaxf(v, 0.f);
}

// ---------------------------------------------------------------------------
extern "C" void kernel_launch(void* const* d_in, const int* in_sizes, int n_in,
                              void* d_out, int out_size, void* d_ws, size_t ws_size,
                              hipStream_t stream) {
  const float* x   = (const float*)d_in[0];
  const float* c0w = (const float*)d_in[1];  const float* c0b = (const float*)d_in[2];
  const float* g0  = (const float*)d_in[3];  const float* b0  = (const float*)d_in[4];
  const float* c1w = (const float*)d_in[5];  const float* c1b = (const float*)d_in[6];
  const float* g1  = (const float*)d_in[7];  const float* b1  = (const float*)d_in[8];
  const float* c2w = (const float*)d_in[9];  const float* c2b = (const float*)d_in[10];
  const float* g2  = (const float*)d_in[11]; const float* b2  = (const float*)d_in[12];
  float* out = (float*)d_out;
  float* ws = (float*)d_ws;

  // scratch planes (each kPlaneTot floats), live inside the conv-ping region
  auto S = [&](int i) { return ws + (long)i * kPlaneTot; };
  float* P0  = ws;                    // conv ping [8,8,512,512] (reuses S0..S7)
  float* P1  = ws + OFF_P1;           // conv pong
  float* XHC = ws + OFF_XHC;          // [8,6,512,512]
  float* ST  = ws + OFF_STATS;
  unsigned* SU = (unsigned*)ST;
  float* PSUM = ST + ST_PSUM;
  float* PSSQ = ST + ST_PSSQ;
  float* SCALE = ST + ST_SCALE;
  float* SHIFT = ST + ST_SHIFT;

  const int EW = kPlaneTot / 256;     // 8192 blocks for per-plane elementwise

  // ---------------- HC block ----------------
  hipLaunchKernelGGL(k_zero_stats, dim3(1), dim3(64), 0, stream, SU);
  hipLaunchKernelGGL(k_gray,  dim3(EW), dim3(256), 0, stream, x, S(0));
  hipLaunchKernelGGL(k_sobel, dim3(EW), dim3(256), 0, stream, S(0), S(1), S(2));
  hipLaunchKernelGGL(k_hblur7, dim3(EW), dim3(256), 0, stream,
                     S(1), S(2), S(3), S(4), S(5));
  hipLaunchKernelGGL(k_vblur7, dim3(EW), dim3(256), 0, stream,
                     S(3), S(4), S(5), S(1), S(2), S(6));
  hipLaunchKernelGGL(k_dettr, dim3(EW), dim3(256), 0, stream,
                     S(1), S(2), S(6), S(4) /*shi_raw*/, S(3) /*har_raw*/, SU);
  hipLaunchKernelGGL(k_blur3, dim3(EW), dim3(256), 0, stream,
                     S(4), S(5) /*shi_b*/, SU, 24, 32, 0);
  hipLaunchKernelGGL(k_blur3, dim3(EW), dim3(256), 0, stream,
                     S(3), S(1) /*har_b*/, SU, 8, 16, 1);
  hipLaunchKernelGGL(k_fast, dim3(EW), dim3(256), 0, stream, S(0), S(2) /*score*/);
  hipLaunchKernelGGL(k_nms,  dim3(EW), dim3(256), 0, stream, S(2), S(3) /*mask*/);
  hipLaunchKernelGGL(k_blur3, dim3(EW), dim3(256), 0, stream,
                     S(3), S(4) /*heat*/, SU, 40, 48, 0);
  hipLaunchKernelGGL(k_xhc, dim3(EW), dim3(256), 0, stream,
                     x, S(1), S(5), S(4), SU, XHC);

  // ---------------- conv + BN + ReLU stack ----------------
  const int CONV_BLOCKS = (kNB * kH * (kW / 16)) / 8;  // 16384
  // layer 0: 6 -> 8, K = 150 -> 5 chunks
  hipLaunchKernelGGL(k_conv_wmma, dim3(CONV_BLOCKS), dim3(256), 0, stream,
                     XHC, c0w, c0b, P0, 6, 5);
  hipLaunchKernelGGL(k_bnstats, dim3(4096), dim3(256), 0, stream, P0, PSUM, PSSQ);
  hipLaunchKernelGGL(k_bnfinal, dim3(1), dim3(32), 0, stream,
                     PSUM, PSSQ, g0, b0, SCALE, SHIFT);
  hipLaunchKernelGGL(k_bnrelu, dim3(65536), dim3(256), 0, stream, P0, SCALE, SHIFT);
  // layer 1: 8 -> 8, K = 200 -> 7 chunks
  hipLaunchKernelGGL(k_conv_wmma, dim3(CONV_BLOCKS), dim3(256), 0, stream,
                     P0, c1w, c1b, P1, 8, 7);
  hipLaunchKernelGGL(k_bnstats, dim3(4096), dim3(256), 0, stream, P1, PSUM, PSSQ);
  hipLaunchKernelGGL(k_bnfinal, dim3(1), dim3(32), 0, stream,
                     PSUM, PSSQ, g1, b1, SCALE, SHIFT);
  hipLaunchKernelGGL(k_bnrelu, dim3(65536), dim3(256), 0, stream, P1, SCALE, SHIFT);
  // layer 2: 8 -> 8, writes d_out
  hipLaunchKernelGGL(k_conv_wmma, dim3(CONV_BLOCKS), dim3(256), 0, stream,
                     P1, c2w, c2b, out, 8, 7);
  hipLaunchKernelGGL(k_bnstats, dim3(4096), dim3(256), 0, stream, out, PSUM, PSSQ);
  hipLaunchKernelGGL(k_bnfinal, dim3(1), dim3(32), 0, stream,
                     PSUM, PSSQ, g2, b2, SCALE, SHIFT);
  hipLaunchKernelGGL(k_bnrelu, dim3(65536), dim3(256), 0, stream, out, SCALE, SHIFT);

  (void)in_sizes; (void)n_in; (void)out_size; (void)ws_size;
}
